// MultiScaleRetention_41661182771657
// MI455X (gfx1250) — matrix operand order repaired and checked by reading, not compiled, not run on hardware
//
#include <hip/hip_runtime.h>
#include <math.h>

#define EMBED   2048
#define NHEADS  16
#define HDIM    128
#define BATCHN  2
#define SEQLEN  2048
#define BNROWS  (BATCHN * SEQLEN)   // 4096

typedef __attribute__((ext_vector_type(16))) __bf16 v16bf;
typedef __attribute__((ext_vector_type(8)))  __bf16 bf16x8;
typedef __attribute__((ext_vector_type(8)))  float  v8f;
typedef __attribute__((ext_vector_type(4)))  float  f32x4;
typedef int v4i __attribute__((vector_size(16)));   // matches builtin prototype

static __device__ __forceinline__ __bf16 f2bf(float f) {
    unsigned u = __builtin_bit_cast(unsigned, f);
    unsigned r = (u + 0x7FFFu + ((u >> 16) & 1u)) >> 16;
    unsigned short s = (unsigned short)r;
    return __builtin_bit_cast(__bf16, s);
}
static __device__ __forceinline__ float bf2f(__bf16 b) {
    unsigned short s = __builtin_bit_cast(unsigned short, b);
    unsigned u = ((unsigned)s) << 16;
    return __builtin_bit_cast(float, u);
}

union AFrag { v16bf v; bf16x8 h2[2]; };

// ---- CDNA5 async global->LDS copy (ASYNCcnt path), with guarded fallback ----
#if defined(__has_builtin)
#if __has_builtin(__builtin_amdgcn_global_load_async_to_lds_b128) && \
    __has_builtin(__builtin_amdgcn_s_wait_asynccnt)
#define HAVE_ASYNC 1
#endif
#endif

static __device__ __forceinline__ void async_copy16(void* lds, const void* g) {
#if defined(HAVE_ASYNC)
    __builtin_amdgcn_global_load_async_to_lds_b128(
        (__attribute__((address_space(1))) v4i*)g,
        (__attribute__((address_space(3))) v4i*)lds, 0, 0);
#else
    *reinterpret_cast<bf16x8*>(lds) = *reinterpret_cast<const bf16x8*>(g);
#endif
}
static __device__ __forceinline__ void async_wait0() {
#if defined(HAVE_ASYNC)
    __builtin_amdgcn_s_wait_asynccnt(0);
#endif
}
static __device__ __forceinline__ void sched_fence() {
#if defined(__has_builtin)
#if __has_builtin(__builtin_amdgcn_sched_barrier)
    __builtin_amdgcn_sched_barrier(0);
#endif
#endif
}

// ------------------------- fp32 -> bf16 convert -------------------------
__global__ __launch_bounds__(256)
void cvt_kernel(const float* __restrict__ src, __bf16* __restrict__ dst, int n8)
{
    int i = blockIdx.x * 256 + threadIdx.x;
    if (i < n8) {
        f32x4 a = *reinterpret_cast<const f32x4*>(src + (size_t)i * 8);
        f32x4 b = *reinterpret_cast<const f32x4*>(src + (size_t)i * 8 + 4);
        bf16x8 o;
        o[0] = f2bf(a[0]); o[1] = f2bf(a[1]); o[2] = f2bf(a[2]); o[3] = f2bf(a[3]);
        o[4] = f2bf(b[0]); o[5] = f2bf(b[1]); o[6] = f2bf(b[2]); o[7] = f2bf(b[3]);
        *reinterpret_cast<bf16x8*>(dst + (size_t)i * 8) = o;
    }
}

// Modes
#define MODE_Q 0
#define MODE_K 1
#define MODE_V 2
#define MODE_G 3
#define MODE_O 4

// C = A @ B^T + bias, A/B bf16, per-mode epilogue.
// 128x128 block tile, BK=32, double-buffered async LDS; 8 waves = 4(M) x 2(N).
template <int MODE>
__global__ __launch_bounds__(256)
void gemm_kernel(const __bf16* __restrict__ A, const __bf16* __restrict__ B,
                 const float* __restrict__ bias,
                 __bf16* __restrict__ outb, float* __restrict__ outf)
{
    __shared__ __attribute__((aligned(16))) __bf16 As[2][128][40];
    __shared__ __attribute__((aligned(16))) __bf16 Bs[2][128][40];

    const int tid  = threadIdx.x;
    const int lane = tid & 31;
    const int wave = tid >> 5;
    const int wm   = wave & 3;
    const int wn   = wave >> 2;
    const int hv   = lane >> 4;
    const int lc   = lane & 15;
    const int mbase = blockIdx.x * 128;
    const int nbase = blockIdx.y * 128;

    v8f acc[2][4];
#pragma unroll
    for (int i = 0; i < 2; ++i)
#pragma unroll
        for (int j = 0; j < 4; ++j)
#pragma unroll
            for (int r = 0; r < 8; ++r) acc[i][j][r] = 0.0f;

    auto stage = [&](int k0, int buf) {
#pragma unroll
        for (int i = 0; i < 2; ++i) {
            int idx = tid + i * 256;            // 0..511 groups of 8
            int r = idx >> 2, cg = (idx & 3) * 8;
            async_copy16(&As[buf][r][cg], A + (size_t)(mbase + r) * EMBED + k0 + cg);
            async_copy16(&Bs[buf][r][cg], B + (size_t)(nbase + r) * EMBED + k0 + cg);
        }
    };

    stage(0, 0);
    for (int kk = 0; kk < EMBED / 32; ++kk) {
        async_wait0();
        __syncthreads();
        if (kk + 1 < EMBED / 32) stage((kk + 1) * 32, (kk + 1) & 1);
        const int buf = kk & 1;

        AFrag a[2];
#pragma unroll
        for (int tm = 0; tm < 2; ++tm) {
            int row = wm * 32 + tm * 16 + lc;       // A layout: lane = M
            a[tm].h2[0] = *reinterpret_cast<const bf16x8*>(&As[buf][row][hv * 8]);
            a[tm].h2[1] = *reinterpret_cast<const bf16x8*>(&As[buf][row][16 + hv * 8]);
        }
        AFrag b[4];
#pragma unroll
        for (int tn = 0; tn < 4; ++tn) {
            int row = wn * 64 + tn * 16 + lc;       // B layout: lane = N
            b[tn].h2[0] = *reinterpret_cast<const bf16x8*>(&Bs[buf][row][hv * 8]);
            b[tn].h2[1] = *reinterpret_cast<const bf16x8*>(&Bs[buf][row][16 + hv * 8]);
        }
        sched_fence();
#pragma unroll
        for (int tm = 0; tm < 2; ++tm)
#pragma unroll
            for (int tn = 0; tn < 4; ++tn)
                acc[tm][tn] = __builtin_amdgcn_wmma_f32_16x16x32_bf16(
                    false, a[tm].v, false, b[tn].v, (short)0, acc[tm][tn],
                    false, false);
        sched_fence();
        __syncthreads();
    }

    // ---- epilogue ----
    const float INV_SQRT_D = 0.08838834764831845f;       // 1/sqrt(128)
    const float L2_10000   = 13.287712379549449f;        // log2(10000)
#pragma unroll
    for (int tm = 0; tm < 2; ++tm) {
#pragma unroll
        for (int tn = 0; tn < 4; ++tn) {
            const int e = nbase + wn * 64 + tn * 16 + lc;    // output feature
            const float bv = bias[e];
            const int h  = e >> 7;
            const int dd = e & (HDIM - 1);
            float theta;
            if constexpr (MODE == MODE_Q || MODE == MODE_K)
                theta = exp2f(-(float)(dd >> 1) * (L2_10000 / 63.0f));
#pragma unroll
            for (int r = 0; r < 8; ++r) {
                const int mg  = mbase + wm * 32 + tm * 16 + r + hv * 8;
                const int seq = mg & (SEQLEN - 1);
                const int bb  = mg >> 11;
                float val = acc[tm][tn][r] + bv;
                if constexpr (MODE == MODE_Q || MODE == MODE_K) {
                    float ang = (float)seq * theta;
                    float sn, cs;
                    __sincosf(ang, &sn, &cs);
                    float p = __shfl_xor(val, 1);        // partner column
                    float rot = (dd & 1) ? (val * cs + p * sn)
                                         : (val * cs - p * sn);
                    if constexpr (MODE == MODE_K) rot *= INV_SQRT_D;
                    size_t o = ((size_t)(bb * NHEADS + h) * SEQLEN + seq) * HDIM + dd;
                    outb[o] = f2bf(rot);
                } else if constexpr (MODE == MODE_V) {
                    // store V transposed: [b, h, d, n]
                    size_t o = ((size_t)(bb * NHEADS + h) * HDIM + dd) * SEQLEN + seq;
                    outb[o] = f2bf(val);
                } else if constexpr (MODE == MODE_G) {
                    float s = val / (1.0f + __expf(-val));   // SiLU
                    outb[(size_t)mg * EMBED + e] = f2bf(s);
                } else {                                      // MODE_O
                    outf[(size_t)mg * EMBED + e] = val;
                }
            }
        }
    }
}

// Retention per (b,h): 64 query rows per block, 4 waves x 16-row strips.
// Async double-buffered key/value chunks; fused decay + group-norm + gate.
__global__ __launch_bounds__(128)
void retention_kernel(const __bf16* __restrict__ Qr, const __bf16* __restrict__ Kr,
                      const __bf16* __restrict__ VtG, const __bf16* __restrict__ gate,
                      __bf16* __restrict__ Y)
{
    __shared__ __attribute__((aligned(16))) __bf16 Klds[2][32][136];
    __shared__ __attribute__((aligned(16))) __bf16 Vt[2][128][40];
    __shared__ __attribute__((aligned(16))) __bf16 Slds[4][16][40];

    const int tid  = threadIdx.x;
    const int lane = tid & 31;
    const int wave = tid >> 5;
    const int hv   = lane >> 4;
    const int lc   = lane & 15;
    const int qblk = blockIdx.x;
    const int bh   = blockIdx.y;
    const int bb   = bh >> 4;
    const int h    = bh & 15;
    const int qbase = qblk * 64;

    const float lg0 = -3.4657359027997265f;   // log(1/32)
    const float lg1 = -6.2383246250395075f;   // log(1/512)
    float gamma = 1.0f - __expf(lg0 + (lg1 - lg0) * ((float)h / 15.0f));
    float log2g = log2f(gamma);

    const size_t headoff  = (size_t)bh * SEQLEN * HDIM;   // for Kr [b,h,n,d]
    const size_t headoffT = (size_t)bh * HDIM * SEQLEN;   // for VtG [b,h,d,n]

    // Persistent Q fragments (A layout), 4 K-chunks of 32
    AFrag qf[4];
    const int qn_lane = qbase + wave * 16 + lc;
    const __bf16* qrow = Qr + headoff + (size_t)qn_lane * HDIM;
#pragma unroll
    for (int c = 0; c < 4; ++c) {
        qf[c].h2[0] = *reinterpret_cast<const bf16x8*>(qrow + c * 32 + hv * 8);
        qf[c].h2[1] = *reinterpret_cast<const bf16x8*>(qrow + c * 32 + 16 + hv * 8);
    }

    v8f acc[8];
#pragma unroll
    for (int i = 0; i < 8; ++i)
#pragma unroll
        for (int r = 0; r < 8; ++r) acc[i][r] = 0.0f;

    auto stage = [&](int kbase, int buf) {
#pragma unroll
        for (int i = 0; i < 4; ++i) {
            int idx = tid + i * 128;                 // 0..511 groups of 8
            { // K chunk, row-major [s][d]
                int r = idx >> 4, cg = (idx & 15) * 8;
                async_copy16(&Klds[buf][r][cg],
                             Kr + headoff + (size_t)(kbase + r) * HDIM + cg);
            }
            { // V^T chunk, [d][s]
                int r = idx >> 2, sg = (idx & 3) * 8;
                async_copy16(&Vt[buf][r][sg],
                             VtG + headoffT + (size_t)r * SEQLEN + kbase + sg);
            }
        }
    };

    const int nchunks = qblk * 2 + 2;
    stage(0, 0);
    for (int ch = 0; ch < nchunks; ++ch) {
        async_wait0();
        __syncthreads();
        if (ch + 1 < nchunks) stage((ch + 1) * 32, (ch + 1) & 1);
        const int buf   = ch & 1;
        const int kbase = ch * 32;

        // S = Q(16x128) x K^T(128x32), decay-masked, -> bf16 per-wave LDS
#pragma unroll
        for (int st = 0; st < 2; ++st) {
            v8f s;
#pragma unroll
            for (int r = 0; r < 8; ++r) s[r] = 0.0f;
            AFrag kb[4];
#pragma unroll
            for (int c = 0; c < 4; ++c) {
                int row = st * 16 + lc;              // B layout: lane = key
                kb[c].h2[0] = *reinterpret_cast<const bf16x8*>(&Klds[buf][row][c * 32 + hv * 8]);
                kb[c].h2[1] = *reinterpret_cast<const bf16x8*>(&Klds[buf][row][c * 32 + 16 + hv * 8]);
            }
            sched_fence();
#pragma unroll
            for (int c = 0; c < 4; ++c)
                s = __builtin_amdgcn_wmma_f32_16x16x32_bf16(
                    false, qf[c].v, false, kb[c].v, (short)0, s, false, false);
            sched_fence();
            const int kn  = kbase + st * 16 + lc;
            const int qn0 = qbase + wave * 16 + hv * 8;
#pragma unroll
            for (int r = 0; r < 8; ++r) {
                int dist = (qn0 + r) - kn;
                float m = (dist < 0) ? 0.0f : exp2f((float)dist * log2g);
                Slds[wave][hv * 8 + r][st * 16 + lc] = f2bf(s[r] * m);
            }
        }
        // Re-layout S (C->A) via per-wave LDS (in-order per wave)
        AFrag sf;
        sf.h2[0] = *reinterpret_cast<const bf16x8*>(&Slds[wave][lc][hv * 8]);
        sf.h2[1] = *reinterpret_cast<const bf16x8*>(&Slds[wave][lc][16 + hv * 8]);
        // R(16x128) += S(16x32) x V(32x128)
        AFrag vf[8];
#pragma unroll
        for (int dt = 0; dt < 8; ++dt) {
            int row = dt * 16 + lc;                  // B layout: lane = d col
            vf[dt].h2[0] = *reinterpret_cast<const bf16x8*>(&Vt[buf][row][hv * 8]);
            vf[dt].h2[1] = *reinterpret_cast<const bf16x8*>(&Vt[buf][row][16 + hv * 8]);
        }
        sched_fence();
#pragma unroll
        for (int dt = 0; dt < 8; ++dt)
            acc[dt] = __builtin_amdgcn_wmma_f32_16x16x32_bf16(
                false, sf.v, false, vf[dt].v, (short)0, acc[dt], false, false);
        sched_fence();
        __syncthreads();
    }

    // Fused group-norm (over d=128 per row) * SiLU gate -> Y bf16 [b,n,e]
#pragma unroll
    for (int r = 0; r < 8; ++r) {
        float sum = 0.0f, ss = 0.0f;
        float vals[8];
#pragma unroll
        for (int dt = 0; dt < 8; ++dt) {
            float v = acc[dt][r];
            vals[dt] = v;
            sum += v;
            ss  += v * v;
        }
#pragma unroll
        for (int off = 1; off < 16; off <<= 1) {     // reduce within half-wave
            sum += __shfl_xor(sum, off);
            ss  += __shfl_xor(ss, off);
        }
        float mean = sum * (1.0f / 128.0f);
        float var  = ss * (1.0f / 128.0f) - mean * mean;
        float rstd = rsqrtf(var + 1e-6f);
        int qn = qbase + wave * 16 + hv * 8 + r;
        size_t rowoff = ((size_t)(bb * SEQLEN + qn)) * EMBED + h * HDIM;
#pragma unroll
        for (int dt = 0; dt < 8; ++dt) {
            int dd = dt * 16 + lc;
            float g = bf2f(gate[rowoff + dd]);
            Y[rowoff + dd] = f2bf((vals[dt] - mean) * rstd * g);
        }
    }
}

extern "C" void kernel_launch(void* const* d_in, const int* in_sizes, int n_in,
                              void* d_out, int out_size, void* d_ws, size_t ws_size,
                              hipStream_t stream)
{
    const float* query = (const float*)d_in[0];
    const float* key   = (const float*)d_in[1];
    const float* value = (const float*)d_in[2];
    const float* Wq = (const float*)d_in[3];
    const float* bq = (const float*)d_in[4];
    const float* Wk = (const float*)d_in[5];
    const float* bk = (const float*)d_in[6];
    const float* Wv = (const float*)d_in[7];
    const float* bv = (const float*)d_in[8];
    const float* Wg = (const float*)d_in[9];
    const float* bg = (const float*)d_in[10];
    const float* Wo = (const float*)d_in[11];
    const float* bo = (const float*)d_in[12];

    char* ws = (char*)d_ws;
    const size_t szX = (size_t)BNROWS * EMBED * sizeof(__bf16);  // 16 MB
    const size_t szW = (size_t)EMBED * EMBED * sizeof(__bf16);   // 8 MB
    __bf16* Aq  = (__bf16*)(ws);                     // bf16 query
    __bf16* Ak  = (__bf16*)(ws + szX);               // bf16 key
    __bf16* Av  = (__bf16*)(ws + 2 * szX);           // bf16 value
    char*   wsw = ws + 3 * szX;
    __bf16* Wqb = (__bf16*)(wsw);
    __bf16* Wkb = (__bf16*)(wsw + szW);
    __bf16* Wvb = (__bf16*)(wsw + 2 * szW);
    __bf16* Wgb = (__bf16*)(wsw + 3 * szW);
    __bf16* Wob = (__bf16*)(wsw + 4 * szW);
    char*   wso = wsw + 5 * szW;
    __bf16* Qr  = (__bf16*)(wso);                    // [b,h,n,d] rope'd Q
    __bf16* Kr  = (__bf16*)(wso + szX);              // [b,h,n,d] rope'd+scaled K
    __bf16* VtG = (__bf16*)(wso + 2 * szX);          // [b,h,d,n] V transposed
    __bf16* Gt  = (__bf16*)(wso + 3 * szX);          // [b,n,e] silu(gate)
    __bf16* Yb  = (__bf16*)(wso + 4 * szX);          // [b,n,e] normed*gate

    const int nX8 = (BNROWS * EMBED) / 8;            // 1,048,576
    const int nW8 = (EMBED * EMBED) / 8;             //   524,288
    cvt_kernel<<<nX8 / 256, 256, 0, stream>>>(query, Aq, nX8);
    cvt_kernel<<<nX8 / 256, 256, 0, stream>>>(key,   Ak, nX8);
    cvt_kernel<<<nX8 / 256, 256, 0, stream>>>(value, Av, nX8);
    cvt_kernel<<<nW8 / 256, 256, 0, stream>>>(Wq, Wqb, nW8);
    cvt_kernel<<<nW8 / 256, 256, 0, stream>>>(Wk, Wkb, nW8);
    cvt_kernel<<<nW8 / 256, 256, 0, stream>>>(Wv, Wvb, nW8);
    cvt_kernel<<<nW8 / 256, 256, 0, stream>>>(Wg, Wgb, nW8);
    cvt_kernel<<<nW8 / 256, 256, 0, stream>>>(Wo, Wob, nW8);

    dim3 gg(BNROWS / 128, EMBED / 128);
    dim3 bk256(256);
    gemm_kernel<MODE_Q><<<gg, bk256, 0, stream>>>(Aq, Wqb, bq, Qr, nullptr);
    gemm_kernel<MODE_K><<<gg, bk256, 0, stream>>>(Ak, Wkb, bk, Kr, nullptr);
    gemm_kernel<MODE_V><<<gg, bk256, 0, stream>>>(Av, Wvb, bv, VtG, nullptr);
    gemm_kernel<MODE_G><<<gg, bk256, 0, stream>>>(Aq, Wgb, bg, Gt, nullptr);

    retention_kernel<<<dim3(SEQLEN / 64, BATCHN * NHEADS), dim3(128), 0, stream>>>(
        Qr, Kr, VtG, Gt, Yb);

    gemm_kernel<MODE_O><<<gg, bk256, 0, stream>>>(Yb, Wob, bo, nullptr,
                                                  (float*)d_out);
}
